// MultiHeadAttention_55422257988287
// MI455X (gfx1250) — matrix operand order repaired and checked by reading, not compiled, not run on hardware
//
#include <hip/hip_runtime.h>
#include <hip/hip_bf16.h>

typedef __bf16 bf16;
typedef __attribute__((ext_vector_type(16))) __bf16 v16bf;
typedef __attribute__((ext_vector_type(8)))  __bf16 bf16x8;
typedef __attribute__((ext_vector_type(4)))  __bf16 bf16x4;
typedef __attribute__((ext_vector_type(8)))  float  v8f;
typedef __attribute__((ext_vector_type(4)))  float  f32x4;
typedef int i32x4 __attribute__((vector_size(16)));
typedef unsigned int u32x4v __attribute__((ext_vector_type(4)));
typedef int i32x8v __attribute__((ext_vector_type(8)));
typedef int i32x4v __attribute__((ext_vector_type(4)));

#define D_MODEL 1024
#define T_SEQ   2048
#define NHEAD   16
#define DHEAD   64
#define BATCH   2

// ---- async global->LDS copy (gfx1250 GLOBAL_LOAD_ASYNC_TO_LDS_B128) ----
#if defined(__has_builtin)
#if __has_builtin(__builtin_amdgcn_global_load_async_to_lds_b128) && \
    __has_builtin(__builtin_amdgcn_s_wait_asynccnt)
#define USE_ASYNC_LDS 1
#endif
#if __has_builtin(__builtin_amdgcn_tensor_load_to_lds) && \
    __has_builtin(__builtin_amdgcn_s_wait_tensorcnt)
#define USE_TDM 1
#endif
#endif
#ifndef USE_ASYNC_LDS
#define USE_ASYNC_LDS 0
#endif
#ifndef USE_TDM
#define USE_TDM 0
#endif

__device__ __forceinline__ void async_copy16(const bf16* g, bf16* l) {
#if USE_ASYNC_LDS
  __builtin_amdgcn_global_load_async_to_lds_b128(
      (__attribute__((address_space(1))) i32x4*)(void*)g,
      (__attribute__((address_space(3))) i32x4*)(void*)l, 0, 0);
#else
  *(bf16x8*)l = *(const bf16x8*)g;
#endif
}

__device__ __forceinline__ void async_join() {
#if USE_ASYNC_LDS
  __builtin_amdgcn_s_wait_asynccnt(0);
#endif
}

#if USE_TDM
// One TENSOR_LOAD_TO_LDS of a 2D tile: tile0 x tile1 elements (bf16), tensor
// row stride stride0 (elements). D# layout per CDNA5 ISA 8.3/8.4; groups 2/3
// are zero (<=2D tensor). 6-arg builtin form (extra v8i group + cpol).
__device__ __forceinline__ void tdm_load_2d(const bf16* gsrc, bf16* ldst,
                                            unsigned td0, unsigned td1,
                                            unsigned stride0,
                                            unsigned tile0, unsigned tile1) {
  const unsigned lds_addr =
      (unsigned)(size_t)(__attribute__((address_space(3))) void*)(void*)ldst;
  const unsigned long long ga = (unsigned long long)(size_t)gsrc;
  u32x4v g0;
  g0[0] = 1u;                                  // count=1, user descriptor
  g0[1] = lds_addr;                            // LDS byte address
  g0[2] = (unsigned)ga;                        // global_addr[31:0]
  g0[3] = (unsigned)((ga >> 32) & 0x01ffffffu) | (2u << 30);  // addr[56:32], type=2
  i32x8v g1;
  g1[0] = (int)(1u << 16);                     // wg_mask=0, data_size=1 (2B)
  g1[1] = (int)((td0 & 0xffffu) << 16);        // tensor_dim0[15:0] @ bits 63:48
  g1[2] = (int)((td0 >> 16) | ((td1 & 0xffffu) << 16));  // td0 hi | td1 lo
  g1[3] = (int)((td1 >> 16) | (tile0 << 16));  // td1 hi | tile_dim0
  g1[4] = (int)(tile1 & 0xffffu);              // tile_dim1; tile_dim2=0
  g1[5] = (int)stride0;                        // tensor_dim0_stride[31:0]
  g1[6] = 0;                                   // stride0 hi, dim1_stride lo
  g1[7] = 0;                                   // dim1_stride hi (unused, 2D)
  i32x4v z4 = {0, 0, 0, 0};
  i32x8v z8 = {0, 0, 0, 0, 0, 0, 0, 0};
  __builtin_amdgcn_tensor_load_to_lds(g0, g1, z4, z4, z8, 0);
}
#endif

__device__ __forceinline__ v16bf combine8(bf16x8 lo, bf16x8 hi) {
  return __builtin_shufflevector(lo, hi, 0, 1, 2, 3, 4, 5, 6, 7,
                                 8, 9, 10, 11, 12, 13, 14, 15);
}

__device__ __forceinline__ v8f wmma_bf16(v16bf a, v16bf b, v8f c) {
  return __builtin_amdgcn_wmma_f32_16x16x32_bf16(false, a, false, b,
                                                 (short)0, c, false, false);
}

// A-matrix fragment (16x32 bf16), ISA 7.12.2 layout: lane&15 = M row,
// lanes 16-31 shift K by +8; halves = {koff..koff+7, koff+16..koff+23}.
__device__ __forceinline__ v16bf load_a_frag(const bf16* p0, int ld, int kbase, int lane) {
  const int m = lane & 15;
  const int koff = kbase + ((lane < 16) ? 0 : 8);
  const bf16* p = p0 + (size_t)m * ld + koff;
  bf16x8 lo = *(const bf16x8*)(p);
  bf16x8 hi = *(const bf16x8*)(p + 16);
  return combine8(lo, hi);
}

// B-matrix fragment (32x16 bf16) from row-major [N][K] storage: lane = N,
// lanes 16-31 shift K by +8; halves = {koff..koff+7, koff+16..koff+23}.
__device__ __forceinline__ v16bf load_b_frag(const bf16* p0, int nbase, int ld,
                                             int kbase, int lane) {
  const int n = nbase + (lane & 15);
  const int koff = kbase + ((lane < 16) ? 0 : 8);
  const bf16* p = p0 + (size_t)n * ld + koff;
  bf16x8 lo = *(const bf16x8*)(p);
  bf16x8 hi = *(const bf16x8*)(p + 16);
  return combine8(lo, hi);
}

// Stage a 128x32 tile into LDS as bf16. fp32 source: b128 loads + cvt.
__device__ __forceinline__ void load_tile_128x32(bf16* lds, const float* src,
                                                 int ld, int tid) {
#pragma unroll
  for (int i = 0; i < 4; ++i) {
    const int idx = tid + i * 256;
    const int row = idx >> 3;
    const int c4 = (idx & 7) << 2;
    f32x4 v = *(const f32x4*)(src + (size_t)row * ld + c4);
    bf16x4 o;
    o[0] = (bf16)v[0]; o[1] = (bf16)v[1]; o[2] = (bf16)v[2]; o[3] = (bf16)v[3];
    *(bf16x4*)(lds + row * 32 + c4) = o;
  }
}

// bf16 source: straight 16B copies.
__device__ __forceinline__ void load_tile_128x32(bf16* lds, const bf16* src,
                                                 int ld, int tid) {
#pragma unroll
  for (int i = 0; i < 2; ++i) {
    const int idx = tid + i * 256;
    const int row = idx >> 2;
    const int c8 = (idx & 3) << 3;
    bf16x8 v = *(const bf16x8*)(src + (size_t)row * ld + c8);
    *(bf16x8*)(lds + row * 32 + c8) = v;
  }
}

// Y = X @ W^T + bias.  X:[M,1024] fp32|bf16, W:[1024,1024] fp32 row-major.
// STORE_MODE: 0 = fp32 row-major [M,1024]
//             1 = bf16 [B,H,T,dh]   (Q, K)
//             2 = bf16 [B,H,dh,T]   (V, pre-transposed for attention PV)
template <typename XT, int STORE_MODE>
__global__ __launch_bounds__(256) void gemm_bias_kernel(
    const XT* __restrict__ X, const float* __restrict__ W,
    const float* __restrict__ bias, void* __restrict__ Yv) {
  __shared__ alignas(16) bf16 lA[128 * 32];
  __shared__ alignas(16) bf16 lB[128 * 32];
  const int tid = threadIdx.x;
  const int lane = tid & 31;
  const int wave = tid >> 5;
  const int wr = wave >> 1;
  const int wc = wave & 1;
  const int m0 = blockIdx.y * 128;
  const int n0 = blockIdx.x * 128;

  v8f acc[2][4];
#pragma unroll
  for (int mi = 0; mi < 2; ++mi)
#pragma unroll
    for (int ni = 0; ni < 4; ++ni)
#pragma unroll
      for (int e = 0; e < 8; ++e) acc[mi][ni][e] = 0.0f;

  const int prow = tid >> 3;           // this thread's staging row
  const int pcol = (tid & 7) << 2;
  for (int k0 = 0; k0 < D_MODEL; k0 += 32) {
    __syncthreads();
    load_tile_128x32(lA, X + (size_t)m0 * D_MODEL + k0, D_MODEL, tid);
    load_tile_128x32(lB, W + (size_t)n0 * D_MODEL + k0, D_MODEL, tid);
    if (k0 + 32 < D_MODEL) {           // prefetch next K-step tiles
      __builtin_prefetch(X + (size_t)(m0 + prow) * D_MODEL + k0 + 32 + pcol, 0, 0);
      __builtin_prefetch(W + (size_t)(n0 + prow) * D_MODEL + k0 + 32 + pcol, 0, 0);
    }
    __syncthreads();
    v16bf a[2], b[4];
#pragma unroll
    for (int mi = 0; mi < 2; ++mi)
      a[mi] = load_a_frag(lA + (wr * 32 + mi * 16) * 32, 32, 0, lane);
#pragma unroll
    for (int ni = 0; ni < 4; ++ni)
      b[ni] = load_b_frag(lB, wc * 64 + ni * 16, 32, 0, lane);
#pragma unroll
    for (int mi = 0; mi < 2; ++mi)
#pragma unroll
      for (int ni = 0; ni < 4; ++ni)
        acc[mi][ni] = wmma_bf16(a[mi], b[ni], acc[mi][ni]);
  }

  const int half8 = (lane < 16) ? 0 : 8;
#pragma unroll
  for (int ni = 0; ni < 4; ++ni) {
    const int n = n0 + wc * 64 + ni * 16 + (lane & 15);
    const float bb = bias[n];
#pragma unroll
    for (int mi = 0; mi < 2; ++mi) {
#pragma unroll
      for (int v = 0; v < 8; ++v) {
        const int m = m0 + wr * 32 + mi * 16 + v + half8;
        const float val = acc[mi][ni][v] + bb;
        const int bidx = m >> 11;
        const int t = m & (T_SEQ - 1);
        const int h = n >> 6;
        const int d = n & (DHEAD - 1);
        if (STORE_MODE == 1) {
          ((bf16*)Yv)[((size_t)(bidx * NHEAD + h) * T_SEQ + t) * DHEAD + d] = (bf16)val;
        } else if (STORE_MODE == 2) {
          ((bf16*)Yv)[((size_t)(bidx * NHEAD + h) * DHEAD + d) * T_SEQ + t] = (bf16)val;
        } else {
          ((float*)Yv)[(size_t)m * D_MODEL + n] = val;
        }
      }
    }
  }
}

// Causal flash attention. Block = (b, h, 128 query rows); 8 waves, each owns
// a 16-row strip. K tile: contiguous 8KB -> async-to-LDS engine. V tile:
// strided 2D tile from [dh][T] -> one Tensor Data Mover descriptor.
__global__ __launch_bounds__(256) void attn_kernel(
    const bf16* __restrict__ Qg, const bf16* __restrict__ Kg,
    const bf16* __restrict__ Vt, bf16* __restrict__ Og) {
  __shared__ alignas(16) bf16 lK[64 * 64];       // [key][dh]  B-frags for QK^T
  __shared__ alignas(16) bf16 lVt[64 * 64];      // [dh][key]  B-frags for PV
  __shared__ alignas(16) bf16 lP[8 * 16 * 64];   // per-wave P staging
  const int tid = threadIdx.x;
  const int lane = tid & 31;
  const int wave = tid >> 5;
  const int half8 = (lane < 16) ? 0 : 8;
  const int q0 = blockIdx.x * 128;
  const int h = blockIdx.y;
  const int b = blockIdx.z;
  const size_t base = (size_t)(b * NHEAD + h) * T_SEQ * DHEAD;
  const int qw = q0 + wave * 16;

  v16bf qf[2];
  qf[0] = load_a_frag(Qg + base + (size_t)qw * DHEAD, DHEAD, 0, lane);
  qf[1] = load_a_frag(Qg + base + (size_t)qw * DHEAD, DHEAD, 32, lane);

  v8f o_acc[4];
  float m_i[8], l_i[8];
#pragma unroll
  for (int ni = 0; ni < 4; ++ni)
#pragma unroll
    for (int e = 0; e < 8; ++e) o_acc[ni][e] = 0.0f;
#pragma unroll
  for (int v = 0; v < 8; ++v) { m_i[v] = -1e30f; l_i[v] = 0.0f; }

  bf16* myP = lP + wave * (16 * 64);
  const int ldrow = tid >> 3;          // staging row for this thread
  const int ldc8 = (tid & 7) << 3;
  const int nk = (q0 >> 6) + 2;        // causal: tiles covering keys <= q0+127
  for (int kt = 0; kt < nk; ++kt) {
    const int kb = kt * 64;
    const bf16* kSrc = Kg + base + (size_t)kb * DHEAD;  // row stride DHEAD
    const bf16* vSrc = Vt + base + kb;                  // row stride T_SEQ
    __syncthreads();
#if USE_TDM
    if (wave == 0)                     // one DMA descriptor for the V tile
      tdm_load_2d(vSrc, lVt, /*td0=*/T_SEQ, /*td1=*/DHEAD, /*stride0=*/T_SEQ,
                  /*tile0=*/64, /*tile1=*/64);
#endif
#pragma unroll
    for (int i = 0; i < 2; ++i) {
      const int row = ldrow + i * 32;
      async_copy16(kSrc + (size_t)row * DHEAD + ldc8, lK + row * 64 + ldc8);
#if !USE_TDM
      async_copy16(vSrc + (size_t)row * T_SEQ + ldc8, lVt + row * 64 + ldc8);
#endif
    }
    if (kt + 1 < nk) {                 // prefetch next tiles
      __builtin_prefetch(kSrc + 64 * DHEAD + (size_t)ldrow * DHEAD + ldc8, 0, 0);
      __builtin_prefetch(vSrc + 64 + (size_t)ldrow * T_SEQ + ldc8, 0, 0);
    }
    async_join();
#if USE_TDM
    if (wave == 0) __builtin_amdgcn_s_wait_tensorcnt(0);
#endif
    __syncthreads();

    // S = Q K^T (contract over dh)
    v8f s[4];
#pragma unroll
    for (int ni = 0; ni < 4; ++ni)
#pragma unroll
      for (int e = 0; e < 8; ++e) s[ni][e] = 0.0f;
#pragma unroll
    for (int ks = 0; ks < 2; ++ks)
#pragma unroll
      for (int ni = 0; ni < 4; ++ni)
        s[ni] = wmma_bf16(qf[ks], load_b_frag(lK, ni * 16, 64, ks * 32, lane), s[ni]);

    // scale + causal mask
#pragma unroll
    for (int ni = 0; ni < 4; ++ni) {
      const int kc = kb + ni * 16 + (lane & 15);
#pragma unroll
      for (int v = 0; v < 8; ++v) {
        float sv = s[ni][v] * 0.125f;
        if (kc > qw + v + half8) sv = -1e30f;
        s[ni][v] = sv;
      }
    }

    // online softmax (row stats redundant across the 16 lanes of each half)
    float tmax[8], tsum[8], alpha[8];
#pragma unroll
    for (int v = 0; v < 8; ++v)
      tmax[v] = fmaxf(fmaxf(s[0][v], s[1][v]), fmaxf(s[2][v], s[3][v]));
#pragma unroll
    for (int off = 1; off < 16; off <<= 1)
#pragma unroll
      for (int v = 0; v < 8; ++v)
        tmax[v] = fmaxf(tmax[v], __shfl_xor(tmax[v], off, 32));
#pragma unroll
    for (int v = 0; v < 8; ++v) {
      const float mn = fmaxf(m_i[v], tmax[v]);
      alpha[v] = __expf(m_i[v] - mn);
      m_i[v] = mn;
      tsum[v] = 0.0f;
    }
#pragma unroll
    for (int ni = 0; ni < 4; ++ni)
#pragma unroll
      for (int v = 0; v < 8; ++v) {
        const float p = __expf(s[ni][v] - m_i[v]);
        s[ni][v] = p;
        tsum[v] += p;
      }
#pragma unroll
    for (int off = 1; off < 16; off <<= 1)
#pragma unroll
      for (int v = 0; v < 8; ++v) tsum[v] += __shfl_xor(tsum[v], off, 32);
#pragma unroll
    for (int v = 0; v < 8; ++v) l_i[v] = l_i[v] * alpha[v] + tsum[v];
#pragma unroll
    for (int ni = 0; ni < 4; ++ni)
#pragma unroll
      for (int v = 0; v < 8; ++v) o_acc[ni][v] *= alpha[v];

    // C-layout f32 -> A-layout bf16 via per-wave LDS staging
#pragma unroll
    for (int ni = 0; ni < 4; ++ni) {
      const int nc = ni * 16 + (lane & 15);
#pragma unroll
      for (int v = 0; v < 8; ++v) myP[(v + half8) * 64 + nc] = (bf16)s[ni][v];
    }
    __syncthreads();  // uniform count; orders the cross-lane LDS round-trip

    // O += P @ V (contract over key = 2x32)
#pragma unroll
    for (int ks = 0; ks < 2; ++ks) {
      v16bf pf = load_a_frag(myP, 64, ks * 32, lane);
#pragma unroll
      for (int ni = 0; ni < 4; ++ni)
        o_acc[ni] = wmma_bf16(pf, load_b_frag(lVt, ni * 16, 64, ks * 32, lane), o_acc[ni]);
    }
  }

  // normalize + write bf16 [B,T,D] (lane <-> consecutive d: coalesced)
#pragma unroll
  for (int ni = 0; ni < 4; ++ni) {
    const int d = ni * 16 + (lane & 15);
#pragma unroll
    for (int v = 0; v < 8; ++v) {
      const int t = qw + v + half8;
      const float ov = o_acc[ni][v] / l_i[v];
      Og[(size_t)(b * T_SEQ + t) * D_MODEL + h * DHEAD + d] = (bf16)ov;
    }
  }
}

extern "C" void kernel_launch(void* const* d_in, const int* in_sizes, int n_in,
                              void* d_out, int out_size, void* d_ws, size_t ws_size,
                              hipStream_t stream) {
  (void)in_sizes; (void)n_in; (void)out_size; (void)ws_size;
  const float* query = (const float*)d_in[0];
  const float* key_  = (const float*)d_in[1];
  const float* value = (const float*)d_in[2];
  const float* q_w = (const float*)d_in[3];
  const float* q_b = (const float*)d_in[4];
  const float* k_w = (const float*)d_in[5];
  const float* k_b = (const float*)d_in[6];
  const float* v_w = (const float*)d_in[7];
  const float* v_b = (const float*)d_in[8];
  const float* o_w = (const float*)d_in[9];
  const float* o_b = (const float*)d_in[10];

  const size_t elems = (size_t)BATCH * T_SEQ * D_MODEL;
  bf16* Qb = (bf16*)d_ws;        // [B,H,T,dh]
  bf16* Kb = Qb + elems;         // [B,H,T,dh]
  bf16* Vb = Kb + elems;         // [B,H,dh,T] (transposed)
  bf16* Ab = Vb + elems;         // [B,T,D]

  dim3 gblk(D_MODEL / 128, (BATCH * T_SEQ) / 128);
  gemm_bias_kernel<float, 1><<<gblk, 256, 0, stream>>>(query, q_w, q_b, (void*)Qb);
  gemm_bias_kernel<float, 1><<<gblk, 256, 0, stream>>>(key_,  k_w, k_b, (void*)Kb);
  gemm_bias_kernel<float, 2><<<gblk, 256, 0, stream>>>(value, v_w, v_b, (void*)Vb);
  attn_kernel<<<dim3(T_SEQ / 128, NHEAD, BATCH), 256, 0, stream>>>(Qb, Kb, Vb, Ab);
  gemm_bias_kernel<bf16, 0><<<gblk, 256, 0, stream>>>(Ab, o_w, o_b, d_out);
}
